// RoIHeadsFast_60301340836224
// MI455X (gfx1250) — matrix operand order repaired and checked by reading, not compiled, not run on hardware
//
#include <hip/hip_runtime.h>
#include <hip/hip_bf16.h>

// ============================================================================
// RoI heads (ROI-align + 4x conv3x3(256->256,ReLU) + conv3x3(256->42))
// Implicit-GEMM via v_wmma_f32_16x16x32_f16, f16 activations/weights, f32 acc.
// Per-RoI input patch staged in LDS via global_load_async_to_lds_b128.
// Workspace layout (needs ~72 MB of d_ws):
//   featT  : NHWC f16 features            [2][100][152][256]
//   actA/B : ping-pong activations f16    [1024][49][256]
//   wpk1..5: packed weights f16           [72][Npad][32]
// ============================================================================

typedef _Float16 v16h __attribute__((ext_vector_type(16)));
typedef _Float16 v8h  __attribute__((ext_vector_type(8)));
typedef float    v8f  __attribute__((ext_vector_type(8)));

#define CIN   256
#define HF    100
#define WF    152
#define BF    2
#define KROI  1024
#define NPIX  49      // 7x7
#define KO    72      // 9 * (256/32) K-chunks of 32

constexpr size_t SZ_FEAT = (size_t)BF * HF * WF * CIN * 2;       // 15,564,800
constexpr size_t SZ_ACT  = (size_t)KROI * NPIX * CIN * 2;        // 25,690,112
constexpr size_t SZ_W    = (size_t)KO * 256 * 32 * 2;            //  1,179,648
constexpr size_t OFF_FEAT = 0;
constexpr size_t OFF_ACTA = OFF_FEAT + SZ_FEAT;
constexpr size_t OFF_ACTB = OFF_ACTA + SZ_ACT;
constexpr size_t OFF_W1   = OFF_ACTB + SZ_ACT;
constexpr size_t OFF_W2   = OFF_W1 + SZ_W;
constexpr size_t OFF_W3   = OFF_W2 + SZ_W;
constexpr size_t OFF_W4   = OFF_W3 + SZ_W;
constexpr size_t OFF_W5   = OFF_W4 + SZ_W;

// ---------------------------------------------------------------- stage 0a:
// NCHW f32 -> NHWC f16 feature transpose (input becomes L2-resident; rereads hit L2)
__global__ __launch_bounds__(256) void nchw_to_nhwc_f16(
    const float* __restrict__ feat, _Float16* __restrict__ featT, int total) {
  int idx = blockIdx.x * 256 + threadIdx.x;
  if (idx >= total) return;
  int c = idx & 255;
  int x = (idx >> 8) % WF;
  int t = (idx >> 8) / WF;
  int y = t % HF;
  int b = t / HF;
  featT[idx] = (_Float16)feat[(((size_t)b * CIN + c) * HF + y) * WF + x];
}

// ---------------------------------------------------------------- stage 0b:
// Pack OIHW f32 weights -> [ko][npad][32] f16 so a wave's B-tile is contiguous.
// K index = (ky*3+kx)*256 + cin ; chunk ko -> fixed (ky,kx), cin block of 32.
__global__ __launch_bounds__(256) void pack_weights(
    const float* __restrict__ W, _Float16* __restrict__ wpk,
    int npad, int cout, int total) {
  int idx = blockIdx.x * 256 + threadIdx.x;
  if (idx >= total) return;
  int k  = idx & 31;
  int n  = (idx >> 5) % npad;
  int ko = (idx >> 5) / npad;
  int kk = ko >> 3;
  int cin = ((ko & 7) << 5) + k;
  int ky = kk / 3, kx = kk - ky * 3;
  float v = 0.f;
  if (n < cout) v = W[(((size_t)n * CIN + cin) * 3 + ky) * 3 + kx];
  wpk[idx] = (_Float16)v;
}

// ---------------------------------------------------------------- stage 1:
// ROI-align, one block per (roi, output pixel), thread = channel.
__global__ __launch_bounds__(256) void roi_align_k(
    const float* __restrict__ rois, const _Float16* __restrict__ featT,
    _Float16* __restrict__ actout) {
  int roi = blockIdx.x, p = blockIdx.y, c = threadIdx.x;
  int oy = p / 7, ox = p % 7;
  const float* r = rois + (size_t)roi * 5;
  int   b  = (int)r[0];
  float x1 = r[1], y1 = r[2], x2 = r[3], y2 = r[4];
  float rw = fmaxf(x2 - x1, 1.f), rh = fmaxf(y2 - y1, 1.f);
  float sum = 0.f;
#pragma unroll
  for (int sy = 0; sy < 2; ++sy)
#pragma unroll
    for (int sx = 0; sx < 2; ++sx) {
      float gy = y1 + ((float)(oy * 2 + sy) + 0.5f) * 0.5f * (rh * (1.f / 7.f));
      float gx = x1 + ((float)(ox * 2 + sx) + 0.5f) * 0.5f * (rw * (1.f / 7.f));
      bool valid = (gy > -1.f) & (gy < (float)HF) & (gx > -1.f) & (gx < (float)WF);
      float yc = fminf(fmaxf(gy, 0.f), (float)(HF - 1));
      float xc = fminf(fmaxf(gx, 0.f), (float)(WF - 1));
      float y0 = fminf(floorf(yc), (float)(HF - 2));
      float x0 = fminf(floorf(xc), (float)(WF - 2));
      float wy = yc - y0, wx = xc - x0;
      const _Float16* base =
          featT + (((size_t)b * HF + (int)y0) * WF + (int)x0) * CIN + c;
      float v00 = (float)base[0];
      float v01 = (float)base[CIN];
      float v10 = (float)base[WF * CIN];
      float v11 = (float)base[WF * CIN + CIN];
      float v = v00 * (1.f - wy) * (1.f - wx) + v01 * (1.f - wy) * wx +
                v10 * wy * (1.f - wx)         + v11 * wy * wx;
      sum += valid ? v : 0.f;
    }
  actout[((size_t)roi * NPIX + p) * CIN + c] = (_Float16)(sum * 0.25f);
}

// ---------------------------------------------------------------- stage 2..6:
// Implicit-GEMM conv3x3 (pad=1) per RoI: O[49 x N] = A[49 x 2304] x Wpk[2304 x N].
// Block: 1 RoI, 256 threads = 8 waves; each wave owns all 4 m-tiles x 2 n-tiles
// (8 f32 accumulators) so one B fragment feeds 4 WMMAs.
// LDS holds the zero-padded 9x9x256 f16 input patch, filled by async-to-LDS copies.
template <bool RELU, bool FINAL, int NT_TOTAL, int NPAD>
__global__ __launch_bounds__(256) void conv3x3_wmma(
    const _Float16* __restrict__ actin,   // [1024][49][256] f16
    const _Float16* __restrict__ wpk,     // [72][NPAD][32]  f16
    const float*    __restrict__ bias,    // [Cout] f32
    _Float16*       __restrict__ actout,  // [1024][49][256] f16   (!FINAL)
    float*          __restrict__ out) {   // [1024][42][49]  f32   (FINAL)
  __shared__ alignas(32) _Float16 tile[81 * CIN];   // 41,472 B

  const int roi = blockIdx.x;
  const int tid = threadIdx.x;

  // ---- Fill padded 9x9x256 patch ----
  {
    // Zero the 32 halo pixels (32 uint4 chunks each).
    const uint4 z = make_uint4(0u, 0u, 0u, 0u);
    uint4* dst = (uint4*)tile;
    for (int i = tid; i < 32 * 32; i += 256) {
      int j = i >> 5, ch = i & 31;
      int py, px;
      if (j < 9)       { py = 0;      px = j;      }
      else if (j < 18) { py = 8;      px = j - 9;  }
      else if (j < 25) { py = j - 17; px = 0;      }
      else             { py = j - 24; px = 8;      }
      dst[(py * 9 + px) * 32 + ch] = z;
    }
    // Interior 7x7x256 via async global->LDS b128 copies (ASYNCcnt path).
    const _Float16* src = actin + (size_t)roi * NPIX * CIN;
    for (int i = tid; i < NPIX * 32; i += 256) {   // 1568 chunks of 16 B
      int p = i >> 5, ch = i & 31;
      int oy = p / 7, ox = p - oy * 7;
      unsigned lds_b = (unsigned)(uintptr_t)&tile[(((oy + 1) * 9 + (ox + 1)) << 8) + ch * 8];
      unsigned goff  = (unsigned)i << 4;           // (p*256 + ch*8) halves * 2 B
      asm volatile("global_load_async_to_lds_b128 %0, %1, %2"
                   :: "v"(lds_b), "v"(goff), "s"(src) : "memory");
    }
    asm volatile("s_wait_asynccnt 0x0" ::: "memory");
  }
  __syncthreads();

  const int wave = tid >> 5, lane = tid & 31;
  const int nt0  = wave * 2;
  if (nt0 >= NT_TOTAL) return;                 // after the barrier: safe
  const bool has1 = ((NT_TOTAL & 1) == 0) ? true : ((nt0 + 1) < NT_TOTAL);

  const int col   = lane & 15;                 // A-row / B-col / C-col for lane
  const int lhalf = lane >> 4;                 // lane half selects K sub-block

  // Per-m-tile A base offsets (pixel of this lane's A-row, clamped for pads).
  int pixoff[4];
#pragma unroll
  for (int m = 0; m < 4; ++m) {
    int pa = m * 16 + col;
    if (pa > 48) pa = 48;
    int oy = pa / 7, ox = pa - oy * 7;
    pixoff[m] = ((oy * 9 + ox) << 8) + lhalf * 16;
  }

  v8f acc[4][2];
#pragma unroll
  for (int m = 0; m < 4; ++m)
#pragma unroll
    for (int j = 0; j < 2; ++j)
      acc[m][j] = (v8f){0.f, 0.f, 0.f, 0.f, 0.f, 0.f, 0.f, 0.f};

  const _Float16* wb0 = wpk + ((size_t)nt0 * 16 + col) * 32 + lhalf * 16;
  const _Float16* wb1 = wb0 + 16 * 32;         // next n-tile

#pragma unroll 2
  for (int ko = 0; ko < KO; ++ko) {
    const int kk = ko >> 3;
    const int cb = (ko & 7) << 5;              // cin base of this K-chunk
    const int ky = kk / 3, kx = kk - ky * 3;
    const int koff = ((ky * 9 + kx) << 8) + cb;
    // B layout used here (same order as A): lane = column, 16 contiguous K halves per lane half.
    v16h b0 = *(const v16h*)&wb0[(size_t)ko * NPAD * 32];
    v16h b1 = has1 ? *(const v16h*)&wb1[(size_t)ko * NPAD * 32] : b0;
#pragma unroll
    for (int m = 0; m < 4; ++m) {
      // A layout: lanes 0-15 hold K [0,16), lanes 16-31 K [16,32) (matches B)
      v8h alo = *(const v8h*)&tile[pixoff[m] + koff];
      v8h ahi = *(const v8h*)&tile[pixoff[m] + koff + 8];
      v16h a = __builtin_shufflevector(alo, ahi, 0, 1, 2, 3, 4, 5, 6, 7,
                                                 8, 9, 10, 11, 12, 13, 14, 15);
      acc[m][0] = __builtin_amdgcn_wmma_f32_16x16x32_f16(
          false, a, false, b0, (short)0, acc[m][0], false, false);
      if (has1)
        acc[m][1] = __builtin_amdgcn_wmma_f32_16x16x32_f16(
            false, a, false, b1, (short)0, acc[m][1], false, false);
    }
  }

  // ---- Epilogue: bias (+ReLU), store f16 NHWC (or f32 NCHW for final) ----
#pragma unroll
  for (int j = 0; j < 2; ++j) {
    if (j == 1 && !has1) break;
    const int n = (nt0 + j) * 16 + col;
    const float bv = (!FINAL || n < 42) ? bias[n] : 0.f;
#pragma unroll
    for (int m = 0; m < 4; ++m) {
#pragma unroll
      for (int i = 0; i < 8; ++i) {
        const int p = m * 16 + i + (lhalf << 3);   // C layout: M = i + 8*(lane/16)
        if (p < NPIX) {
          float v = acc[m][j][i] + bv;
          if (RELU) v = v > 0.f ? v : 0.f;
          if (FINAL) {
            if (n < 42) out[((size_t)roi * 42 + n) * NPIX + p] = v;
          } else {
            actout[((size_t)roi * NPIX + p) * CIN + n] = (_Float16)v;
          }
        }
      }
    }
  }
}

// ============================================================================
extern "C" void kernel_launch(void* const* d_in, const int* in_sizes, int n_in,
                              void* d_out, int out_size, void* d_ws, size_t ws_size,
                              hipStream_t stream) {
  const float* features = (const float*)d_in[0];
  const float* rois     = (const float*)d_in[1];
  const float* W1 = (const float*)d_in[2];  const float* b1 = (const float*)d_in[3];
  const float* W2 = (const float*)d_in[4];  const float* b2 = (const float*)d_in[5];
  const float* W3 = (const float*)d_in[6];  const float* b3 = (const float*)d_in[7];
  const float* W4 = (const float*)d_in[8];  const float* b4 = (const float*)d_in[9];
  const float* W5 = (const float*)d_in[10]; const float* b5 = (const float*)d_in[11];

  char* ws = (char*)d_ws;
  _Float16* featT = (_Float16*)(ws + OFF_FEAT);
  _Float16* actA  = (_Float16*)(ws + OFF_ACTA);
  _Float16* actB  = (_Float16*)(ws + OFF_ACTB);
  _Float16* wp1   = (_Float16*)(ws + OFF_W1);
  _Float16* wp2   = (_Float16*)(ws + OFF_W2);
  _Float16* wp3   = (_Float16*)(ws + OFF_W3);
  _Float16* wp4   = (_Float16*)(ws + OFF_W4);
  _Float16* wp5   = (_Float16*)(ws + OFF_W5);

  // Stage 0: feature transpose + weight packing
  {
    int tot = BF * HF * WF * CIN;
    nchw_to_nhwc_f16<<<(tot + 255) / 256, 256, 0, stream>>>(features, featT, tot);
    int totw = KO * 256 * 32;
    pack_weights<<<(totw + 255) / 256, 256, 0, stream>>>(W1, wp1, 256, 256, totw);
    pack_weights<<<(totw + 255) / 256, 256, 0, stream>>>(W2, wp2, 256, 256, totw);
    pack_weights<<<(totw + 255) / 256, 256, 0, stream>>>(W3, wp3, 256, 256, totw);
    pack_weights<<<(totw + 255) / 256, 256, 0, stream>>>(W4, wp4, 256, 256, totw);
    int totw5 = KO * 48 * 32;
    pack_weights<<<(totw5 + 255) / 256, 256, 0, stream>>>(W5, wp5, 48, 42, totw5);
  }

  // Stage 1: ROI-align -> actA
  roi_align_k<<<dim3(KROI, NPIX), 256, 0, stream>>>(rois, featT, actA);

  // Stage 2-6: conv tower (ping-pong activations), final layer to d_out (f32 NCHW)
  conv3x3_wmma<true,  false, 16, 256><<<dim3(KROI), 256, 0, stream>>>(actA, wp1, b1, actB, nullptr);
  conv3x3_wmma<true,  false, 16, 256><<<dim3(KROI), 256, 0, stream>>>(actB, wp2, b2, actA, nullptr);
  conv3x3_wmma<true,  false, 16, 256><<<dim3(KROI), 256, 0, stream>>>(actA, wp3, b3, actB, nullptr);
  conv3x3_wmma<true,  false, 16, 256><<<dim3(KROI), 256, 0, stream>>>(actB, wp4, b4, actA, nullptr);
  conv3x3_wmma<false, true,  3,  48 ><<<dim3(KROI), 256, 0, stream>>>(actA, wp5, b5, nullptr, (float*)d_out);
}
